// Based_41205916238374
// MI455X (gfx1250) — compile-verified, matrix operand order checked by reading, not run on hardware
//
#include <hip/hip_runtime.h>

// ---------------------------------------------------------------------------
// Types for CDNA5 WMMA (wave32, v_wmma_f32_16x16x32_bf16)
// ---------------------------------------------------------------------------
typedef __bf16 bf16;
typedef __attribute__((ext_vector_type(16))) __bf16 v16bf;
typedef __attribute__((ext_vector_type(8)))  __bf16 v8bf;
typedef __attribute__((ext_vector_type(8)))  float  v8f;

// 16-bit A/B fragment layout (CDNA5 ISA 7.12.2):
//   lane row/col = lane&15 ; K(j) = kbase + j (+8 for j>=8), kbase = kstep*32 + (lane<16?0:8)
// -> two contiguous 16-byte loads per lane.
__device__ __forceinline__ v16bf load_frag(const bf16* p) {
    v8bf lo = *(const v8bf*)p;
    v8bf hi = *(const v8bf*)(p + 16);
    v16bf f;
#pragma unroll
    for (int i = 0; i < 8; ++i) { f[i] = lo[i]; f[i + 8] = hi[i]; }
    return f;
}

__device__ __forceinline__ v8f wmma_bf16(v16bf a, v16bf b, v8f c) {
    return __builtin_amdgcn_wmma_f32_16x16x32_bf16(false, a, false, b, (short)0, c,
                                                   false, false);
}

#define NEG_INF (-__builtin_inff())

// ---------------------------------------------------------------------------
// Conversion / packing kernels
// ---------------------------------------------------------------------------

// src: rows x K (f32) -> dst: rows x Kp (bf16), zero pad K..Kp
__global__ void pad_rows_bf16(const float* __restrict__ src, bf16* __restrict__ dst,
                              long rows, int K, int Kp) {
    long total = rows * (long)Kp;
    for (long idx = (long)blockIdx.x * blockDim.x + threadIdx.x; idx < total;
         idx += (long)gridDim.x * blockDim.x) {
        long r = idx / Kp;
        int  k = (int)(idx % Kp);
        dst[idx] = (k < K) ? (bf16)src[r * K + k] : (bf16)0.0f;
    }
}

// src: K x N (f32, torch-transposed linear weight) -> dst: Np x Kp (bf16), transposed+padded
__global__ void transpose_pad_w(const float* __restrict__ src, bf16* __restrict__ dst,
                                int K, int N, int Kp, int Np) {
    long total = (long)Np * Kp;
    for (long idx = (long)blockIdx.x * blockDim.x + threadIdx.x; idx < total;
         idx += (long)gridDim.x * blockDim.x) {
        int n = (int)(idx / Kp);
        int k = (int)(idx % Kp);
        float v = (n < N && k < K) ? src[(long)k * N + n] : 0.0f;
        dst[idx] = (bf16)v;
    }
}

// src: (B,S,16*dsrc) f32 -> dst: (B,16,S,dpad) bf16 (head-major, zero padded dims)
__global__ void pack_heads(const float* __restrict__ src, bf16* __restrict__ dst,
                           int dsrc, int dpad, long total) {
    for (long idx = (long)blockIdx.x * blockDim.x + threadIdx.x; idx < total;
         idx += (long)gridDim.x * blockDim.x) {
        int  e    = (int)(idx % dpad);
        long rest = idx / dpad;
        int  s    = (int)(rest % 2048); rest /= 2048;
        int  h    = (int)(rest % 16);
        int  b    = (int)(rest / 16);
        float v = (e < dsrc) ? src[((long)(b * 2048 + s)) * 528 + h * dsrc + e] : 0.0f;
        dst[idx] = (bf16)v;
    }
}

// ---------------------------------------------------------------------------
// Generic bf16 WMMA GEMM: C(MxN,f32) = A(MxKp,bf16) @ Bt(NpxKp,bf16)^T [+ Cin]
// Block: 256 thr = 8 waves, block tile 64x64, wave tile 16x32 (A frag reused).
// ---------------------------------------------------------------------------
__global__ void __launch_bounds__(256)
gemm_bf16_wmma(const bf16* __restrict__ A, const bf16* __restrict__ Bt,
               float* __restrict__ C, const float* __restrict__ Cin,
               int M, int N, int Kp) {
    const int tid   = threadIdx.x;
    const int wave  = tid >> 5;
    const int lane  = tid & 31;
    const int lhalf = (lane < 16) ? 0 : 8;
    const int lr    = lane & 15;

    const int row0 = blockIdx.x * 64 + (wave >> 1) * 16;
    const int col0 = blockIdx.y * 64 + (wave & 1) * 32;

    v8f c0 = {};
    v8f c1 = {};
    for (int k = 0; k < Kp; k += 32) {
        const int kb = k + lhalf;
        v16bf a  = load_frag(A  + (size_t)(row0 + lr) * Kp + kb);
        v16bf b0 = load_frag(Bt + (size_t)(col0 + lr) * Kp + kb);
        v16bf b1 = load_frag(Bt + (size_t)(col0 + 16 + lr) * Kp + kb);
        c0 = wmma_bf16(a, b0, c0);
        c1 = wmma_bf16(a, b1, c1);
    }
#pragma unroll
    for (int i = 0; i < 8; ++i) {
        const int r = row0 + i + lhalf;
        const int n = col0 + lr;
        if (r < M) {
            if (n < N) {
                float v = c0[i];
                if (Cin) v += Cin[(size_t)r * N + n];
                C[(size_t)r * N + n] = v;
            }
            if (n + 16 < N) {
                float v = c1[i];
                if (Cin) v += Cin[(size_t)r * N + n + 16];
                C[(size_t)r * N + n + 16] = v;
            }
        }
    }
}

// ---------------------------------------------------------------------------
// Linear (Taylor feature-map) attention; one block per (b,h). Tiny FLOP count,
// VALU fp32 for fidelity. kv[d][e] = sum_n ke[n][d] v[n][e]; out = qe@kv / z.
// ---------------------------------------------------------------------------
#define LIN_JOBS 5
__global__ void __launch_bounds__(256)
linear_attn_kernel(const float* __restrict__ qf, const float* __restrict__ kf,
                   const float* __restrict__ vf, const int* __restrict__ mask,
                   float* __restrict__ out) {
    const int b   = blockIdx.x >> 4;
    const int h   = blockIdx.x & 15;
    const int tid = threadIdx.x;

    __shared__ float kraw[64][17];
    __shared__ float vtile[64][34];
    __shared__ float kv[33][34];
    __shared__ float ksum[33];

    float acc[LIN_JOBS];
#pragma unroll
    for (int j = 0; j < LIN_JOBS; ++j) acc[j] = 0.0f;

    for (int t = 0; t < 2048 / 64; ++t) {
        const int nb = t * 64;
        __syncthreads();
        for (int idx = tid; idx < 64 * 16; idx += 256) {
            int n = idx >> 4, f = idx & 15;
            kraw[n][f] = kf[((size_t)(b * 2048 + nb + n)) * 256 + h * 16 + f];
        }
        for (int idx = tid; idx < 64 * 33; idx += 256) {
            int n = idx / 33, e = idx % 33;
            vtile[n][e] = vf[((size_t)(b * 2048 + nb + n)) * 528 + h * 33 + e];
        }
        __syncthreads();
        for (int j = 0; j < LIN_JOBS; ++j) {
            int p = tid + 256 * j;
            if (p < 33 * 33) {
                int d = p / 33, e = p % 33;
                float s = 0.0f;
                for (int n = 0; n < 64; ++n) {
                    float kd = (d == 0) ? 1.0f
                             : (d <= 16 ? kraw[n][d - 1]
                                        : 0.5f * kraw[n][d - 17] * kraw[n][d - 17]);
                    s += kd * vtile[n][e];
                }
                acc[j] += s;
            } else if (p < 33 * 33 + 33) {
                int e = p - 33 * 33;
                float s = 0.0f;
                for (int n = 0; n < 64; ++n) {
                    float kd = (e == 0) ? 1.0f
                             : (e <= 16 ? kraw[n][e - 1]
                                        : 0.5f * kraw[n][e - 17] * kraw[n][e - 17]);
                    s += kd;
                }
                acc[j] += s;
            }
        }
    }
    __syncthreads();
    for (int j = 0; j < LIN_JOBS; ++j) {
        int p = tid + 256 * j;
        if (p < 33 * 33)            kv[p / 33][p % 33] = acc[j];
        else if (p < 33 * 33 + 33)  ksum[p - 33 * 33]  = acc[j];
    }
    __syncthreads();

    for (int n = tid; n < 2048; n += 256) {
        float qe[33];
        qe[0] = 1.0f;
        float qsum = 1.0f;
#pragma unroll
        for (int f = 0; f < 16; ++f) {
            float q = qf[((size_t)(b * 2048 + n)) * 256 + h * 16 + f];
            qe[1 + f]  = q;
            qe[17 + f] = 0.5f * q * q;
            qsum += qe[1 + f] + qe[17 + f];
        }
        const int mk = mask[b * 2048 + n];
        for (int e = 0; e < 33; ++e) {
            float s = 0.0f;
            for (int d = 0; d < 33; ++d) s += qe[d] * kv[d][e];
            float z = qsum * ksum[e];
            float o = s / (z + 1e-9f);
            out[((size_t)(b * 2048 + n)) * 528 + h * 33 + e] = mk ? o : 0.0f;
        }
    }
}

// ---------------------------------------------------------------------------
// Window (causal + 63 lookahead) attention, flash style with WMMA scores & PV.
// One block per (b, h, 64-row tile). Q/K padded to 64 dims, V padded to 48.
// ---------------------------------------------------------------------------
__global__ void __launch_bounds__(256)
window_attn_kernel(const bf16* __restrict__ qb, const bf16* __restrict__ kb,
                   const bf16* __restrict__ vb, const int* __restrict__ mask,
                   float* __restrict__ out) {
    const int gid = blockIdx.x;
    const int it  = gid & 31;
    const int h   = (gid >> 5) & 15;
    const int b   = gid >> 9;
    const int n0  = it * 64;

    const int tid   = threadIdx.x;
    const int wave  = tid >> 5;
    const int lane  = tid & 31;
    const int lhalf = (lane < 16) ? 0 : 8;
    const int lr    = lane & 15;

    __shared__ __align__(16) bf16 Qs[64 * 64];
    __shared__ __align__(16) bf16 Ks[64 * 64];
    __shared__ __align__(16) bf16 VsT[48 * 64];   // [dim][key] -> contiguous K for B frag
    __shared__ __align__(16) bf16 Ps[64 * 64];
    __shared__ float Ss[64][68];
    __shared__ float Oacc[64][48];
    __shared__ float alph[64];
    __shared__ float lrow[64];

    const size_t bh = (size_t)(b * 16 + h) * 2048;

    for (int idx = tid; idx < 64 * 64; idx += 256)
        Qs[idx] = qb[(bh + n0 + (idx >> 6)) * 64 + (idx & 63)];
    for (int idx = tid; idx < 64 * 48; idx += 256)
        Oacc[idx / 48][idx % 48] = 0.0f;

    float mi = NEG_INF, li = 0.0f;          // row state, valid for tid < 64
    const float scale = 0.17407765595569785f; // 1/sqrt(33)

    int jend = (n0 + 126) >> 6;
    if (jend > 31) jend = 31;

    for (int j = 0; j <= jend; ++j) {
        const int m0 = j * 64;
        __syncthreads();
        for (int idx = tid; idx < 64 * 64; idx += 256)
            Ks[idx] = kb[(bh + m0 + (idx >> 6)) * 64 + (idx & 63)];
        for (int idx = tid; idx < 48 * 64; idx += 256) {
            int d = idx >> 6, key = idx & 63;
            VsT[d * 64 + key] = vb[(bh + m0 + key) * 48 + d];
        }
        __syncthreads();

        // ---- scores: 16 tiles of 16x16, 8 waves x 2
        for (int tt = 0; tt < 2; ++tt) {
            const int t  = wave + tt * 8;
            const int rt = t >> 2, ct = t & 3;
            v8f c = {};
#pragma unroll
            for (int ks = 0; ks < 64; ks += 32) {
                const int kbase = ks + lhalf;
                v16bf a  = load_frag(Qs + (rt * 16 + lr) * 64 + kbase);
                v16bf bm = load_frag(Ks + (ct * 16 + lr) * 64 + kbase);
                c = wmma_bf16(a, bm, c);
            }
            const int  mg  = m0 + ct * 16 + lr;
            const bool mok = mask[b * 2048 + mg] != 0;
#pragma unroll
            for (int i = 0; i < 8; ++i) {
                const int  r  = rt * 16 + i + lhalf;
                const int  ng = n0 + r;
                const bool ok = mok && (mg <= ng + 63);
                Ss[r][ct * 16 + lr] = ok ? c[i] * scale : NEG_INF;
            }
        }
        __syncthreads();

        // ---- online softmax, one thread per row
        if (tid < 64) {
            const int r = tid;
            float mx = NEG_INF;
            for (int cc = 0; cc < 64; ++cc) mx = fmaxf(mx, Ss[r][cc]);
            const float mnew = fmaxf(mi, mx);
            const float mm   = (mnew == NEG_INF) ? 0.0f : mnew;
            const float a    = (mi == NEG_INF) ? 0.0f : __expf(mi - mnew);
            float ps = 0.0f;
            for (int cc = 0; cc < 64; ++cc) {
                float s = Ss[r][cc];
                float p = (s == NEG_INF) ? 0.0f : __expf(s - mm);
                Ps[r * 64 + cc] = (bf16)p;
                ps += p;
            }
            li = li * a + ps;
            mi = mnew;
            alph[r] = a;
        }
        __syncthreads();

        // ---- O = O*alpha + P@V : 12 tiles (4 row x 3 dim), waves 0..7 then 0..3
        for (int tt = 0; tt < 2; ++tt) {
            const int t = wave + tt * 8;
            if (t >= 12) continue;          // wave-uniform: EXEC stays all-ones
            const int rt = t / 3, et = t % 3;
            v8f c;
#pragma unroll
            for (int i = 0; i < 8; ++i) {
                const int r = rt * 16 + i + lhalf;
                c[i] = Oacc[r][et * 16 + lr] * alph[r];
            }
#pragma unroll
            for (int ks = 0; ks < 64; ks += 32) {
                const int kbase = ks + lhalf;
                v16bf a  = load_frag(Ps  + (rt * 16 + lr) * 64 + kbase);
                v16bf bm = load_frag(VsT + (et * 16 + lr) * 64 + kbase);
                c = wmma_bf16(a, bm, c);
            }
#pragma unroll
            for (int i = 0; i < 8; ++i) {
                const int r = rt * 16 + i + lhalf;
                Oacc[r][et * 16 + lr] = c[i];
            }
        }
    }
    if (tid < 64) lrow[tid] = li;
    __syncthreads();

    for (int idx = tid; idx < 64 * 33; idx += 256) {
        const int r = idx / 33, e = idx % 33;
        const float l = lrow[r];
        out[((size_t)(b * 2048) + n0 + r) * 528 + h * 33 + e] =
            (l > 0.0f) ? Oacc[r][e] / l : 0.0f;
    }
}

// ---------------------------------------------------------------------------
// Host orchestration
// ---------------------------------------------------------------------------
static inline dim3 gs(long total) {
    long bl = (total + 255) / 256;
    if (bl > 65535) bl = 65535;
    return dim3((unsigned)bl);
}

extern "C" void kernel_launch(void* const* d_in, const int* in_sizes, int n_in,
                              void* d_out, int out_size, void* d_ws, size_t ws_size,
                              hipStream_t stream) {
    (void)in_sizes; (void)n_in; (void)out_size; (void)ws_size;

    const float* x    = (const float*)d_in[0];
    const int*   mask = (const int*)d_in[1];
    const float* WqL  = (const float*)d_in[2];
    const float* WkL  = (const float*)d_in[3];
    const float* WvL  = (const float*)d_in[4];
    const float* WoL  = (const float*)d_in[5];
    const float* WqW  = (const float*)d_in[6];
    const float* WkW  = (const float*)d_in[7];
    const float* WvW  = (const float*)d_in[8];
    const float* WoW  = (const float*)d_in[9];
    float* out = (float*)d_out;

    const int B = 2, S = 2048, D = 528, Kp = 544, M = B * S; // 4096

    size_t off = 0;
    auto alloc = [&](size_t bytes) -> char* {
        char* p = (char*)d_ws + off;
        off += (bytes + 255) & ~(size_t)255;
        return p;
    };

    bf16* Xb    = (bf16*)alloc((size_t)M * Kp * 2);
    bf16* WqLt  = (bf16*)alloc((size_t)256 * Kp * 2);
    bf16* WkLt  = (bf16*)alloc((size_t)256 * Kp * 2);
    bf16* WvLt  = (bf16*)alloc((size_t)576 * Kp * 2);
    bf16* WoLt  = (bf16*)alloc((size_t)576 * Kp * 2);
    bf16* WqWt  = (bf16*)alloc((size_t)576 * Kp * 2);
    bf16* WkWt  = (bf16*)alloc((size_t)576 * Kp * 2);
    bf16* WvWt  = (bf16*)alloc((size_t)576 * Kp * 2);
    bf16* WoWt  = (bf16*)alloc((size_t)576 * Kp * 2);
    float* q_lin = (float*)alloc((size_t)M * 256 * 4);
    float* k_lin = (float*)alloc((size_t)M * 256 * 4);
    float* v_lin = (float*)alloc((size_t)M * 528 * 4);
    float* q_win = (float*)alloc((size_t)M * 528 * 4);
    float* k_win = (float*)alloc((size_t)M * 528 * 4);
    float* v_win = (float*)alloc((size_t)M * 528 * 4);
    bf16* qbh   = (bf16*)alloc((size_t)B * 16 * S * 64 * 2);
    bf16* kbh   = (bf16*)alloc((size_t)B * 16 * S * 64 * 2);
    bf16* vbh   = (bf16*)alloc((size_t)B * 16 * S * 48 * 2);
    float* lin_pre = (float*)alloc((size_t)M * 528 * 4);
    float* win_pre = (float*)alloc((size_t)M * 528 * 4);
    bf16* linb  = (bf16*)alloc((size_t)M * Kp * 2);
    bf16* winb  = (bf16*)alloc((size_t)M * Kp * 2);

    // 1) pack activations + weights to bf16 (transposed, K-padded to 544)
    pad_rows_bf16<<<gs((long)M * Kp), 256, 0, stream>>>(x, Xb, M, D, Kp);
    transpose_pad_w<<<gs((long)256 * Kp), 256, 0, stream>>>(WqL, WqLt, 528, 256, Kp, 256);
    transpose_pad_w<<<gs((long)256 * Kp), 256, 0, stream>>>(WkL, WkLt, 528, 256, Kp, 256);
    transpose_pad_w<<<gs((long)576 * Kp), 256, 0, stream>>>(WvL, WvLt, 528, 528, Kp, 576);
    transpose_pad_w<<<gs((long)576 * Kp), 256, 0, stream>>>(WoL, WoLt, 528, 528, Kp, 576);
    transpose_pad_w<<<gs((long)576 * Kp), 256, 0, stream>>>(WqW, WqWt, 528, 528, Kp, 576);
    transpose_pad_w<<<gs((long)576 * Kp), 256, 0, stream>>>(WkW, WkWt, 528, 528, Kp, 576);
    transpose_pad_w<<<gs((long)576 * Kp), 256, 0, stream>>>(WvW, WvWt, 528, 528, Kp, 576);
    transpose_pad_w<<<gs((long)576 * Kp), 256, 0, stream>>>(WoW, WoWt, 528, 528, Kp, 576);

    // 2) six input projections (WMMA GEMMs)
    dim3 blk(256);
    dim3 g256(M / 64, (256 + 63) / 64);
    dim3 g528(M / 64, (528 + 63) / 64);
    gemm_bf16_wmma<<<g256, blk, 0, stream>>>(Xb, WqLt, q_lin, nullptr, M, 256, Kp);
    gemm_bf16_wmma<<<g256, blk, 0, stream>>>(Xb, WkLt, k_lin, nullptr, M, 256, Kp);
    gemm_bf16_wmma<<<g528, blk, 0, stream>>>(Xb, WvLt, v_lin, nullptr, M, 528, Kp);
    gemm_bf16_wmma<<<g528, blk, 0, stream>>>(Xb, WqWt, q_win, nullptr, M, 528, Kp);
    gemm_bf16_wmma<<<g528, blk, 0, stream>>>(Xb, WkWt, k_win, nullptr, M, 528, Kp);
    gemm_bf16_wmma<<<g528, blk, 0, stream>>>(Xb, WvWt, v_win, nullptr, M, 528, Kp);

    // 3) repack window q/k/v per-head (bf16, dim-padded 64/64/48)
    pack_heads<<<gs((long)B * 16 * S * 64), 256, 0, stream>>>(q_win, qbh, 33, 64,
                                                              (long)B * 16 * S * 64);
    pack_heads<<<gs((long)B * 16 * S * 64), 256, 0, stream>>>(k_win, kbh, 33, 64,
                                                              (long)B * 16 * S * 64);
    pack_heads<<<gs((long)B * 16 * S * 48), 256, 0, stream>>>(v_win, vbh, 33, 48,
                                                              (long)B * 16 * S * 48);

    // 4) attention bodies
    linear_attn_kernel<<<dim3(B * 16), blk, 0, stream>>>(q_lin, k_lin, v_lin, mask, lin_pre);
    window_attn_kernel<<<dim3(B * 16 * (S / 64)), blk, 0, stream>>>(qbh, kbh, vbh, mask,
                                                                    win_pre);

    // 5) output projections: out = lin_pre @ WoL + win_pre @ WoW
    pad_rows_bf16<<<gs((long)M * Kp), 256, 0, stream>>>(lin_pre, linb, M, D, Kp);
    pad_rows_bf16<<<gs((long)M * Kp), 256, 0, stream>>>(win_pre, winb, M, D, Kp);
    gemm_bf16_wmma<<<g528, blk, 0, stream>>>(linb, WoLt, out, nullptr, M, 528, Kp);
    gemm_bf16_wmma<<<g528, blk, 0, stream>>>(winb, WoWt, out, out, M, 528, Kp);
}
